// DiMBlock_15247133901494
// MI455X (gfx1250) — matrix operand order repaired and checked by reading, not compiled, not run on hardware
//
#include <hip/hip_runtime.h>

// ---- problem constants (from reference) ----
#define BB     8
#define LL     1024
#define DM     1024
#define DI     2048
#define DSTATE 2
#define DTRN   64
#define MLPH   4096
#define MROWS  (BB*LL)    // 8192
#define MODS   6144
#define XPN    (DTRN + 2*DSTATE)   // 68

typedef __attribute__((ext_vector_type(16))) __bf16 v16bf;
typedef __attribute__((ext_vector_type(8)))  float  v8f;

__device__ __forceinline__ float siluf(float x) { return x / (1.0f + __expf(-x)); }

// gfx1250 async copy: global -> LDS, 16 bytes per lane, tracked by ASYNCcnt.
__device__ __forceinline__ void async_ld_b128(unsigned lds_off, const float* gptr) {
    asm volatile("global_load_async_to_lds_b128 %0, %1, off"
                 :: "v"(lds_off), "v"((unsigned long long)(size_t)gptr)
                 : "memory");
}
__device__ __forceinline__ void wait_async0() {
    asm volatile("s_wait_asynccnt 0x0" ::: "memory");
}

// =====================================================================
// WMMA GEMM: Out[M,NN] = epi( A[M,KK] @ W[NN,KK]^T ), M = 8192.
// block = 256 threads (8 waves), tile 128(M) x 64(N), K-step 32.
// fp32 tiles double-buffered in LDS via GLOBAL_LOAD_ASYNC_TO_LDS_B128;
// bf16 conversion fused into fragment build; 2x2 WMMA tiles per wave.
// EPI: 0 = store raw; 1 = bias + tanh-GELU; 2 = resid + gate*(acc+bias)
// REV: reverse sequence within each batch when reading A rows (ssm2 path).
// =====================================================================
template <int EPI, bool REV, int NN, int KK>
__global__ __launch_bounds__(256) void gemm_bf16_wmma(
    const float* __restrict__ A, const float* __restrict__ W,
    const float* __restrict__ bias, const float* __restrict__ resid,
    const float* __restrict__ gate,   // mods + 5*DM (stride MODS) for EPI==2
    float* __restrict__ Out)
{
    constexpr int LDT = 36;                 // padded fp32 LDS row stride
    __shared__ __align__(16) float sA[2][128 * LDT];
    __shared__ __align__(16) float sB[2][64 * LDT];

    const int m0 = blockIdx.y * 128;
    const int n0 = blockIdx.x * 64;
    const int t = threadIdx.x;
    const int lane = t & 31;
    const int wave = t >> 5;
    const int wm = wave >> 1;   // 0..3
    const int wn = wave & 1;    // 0..1
    const int r  = lane & 15;
    const int kh = lane >> 4;

    v8f acc[2][2];
#pragma unroll
    for (int i = 0; i < 2; i++)
#pragma unroll
        for (int j = 0; j < 2; j++) {
            v8f z = {0.f,0.f,0.f,0.f,0.f,0.f,0.f,0.f};
            acc[i][j] = z;
        }

    // cooperative async-load mapping: A 128x32 fp32 (64B/thread), W 64x32 (32B/thread)
    const int arow = t >> 1;          // 0..127
    const int ac   = (t & 1) * 16;    // 0 or 16
    int gar = m0 + arow;
    if (REV) { int b = gar >> 10, l = gar & 1023; gar = (b << 10) + (1023 - l); }
    const float* aptr = A + (size_t)gar * KK + ac;

    const int brow = t >> 2;          // 0..63
    const int bc   = (t & 3) * 8;     // 0,8,16,24
    const float* wptr = W + (size_t)(n0 + brow) * KK + bc;

    unsigned albase[2], blbase[2];
#pragma unroll
    for (int bf = 0; bf < 2; bf++) {
        albase[bf] = (unsigned)(size_t)&sA[bf][arow * LDT + ac];
        blbase[bf] = (unsigned)(size_t)&sB[bf][brow * LDT + bc];
    }

    auto issue = [&](int kt, int buf) {
        const float* ap = aptr + kt * 32;
#pragma unroll
        for (int i = 0; i < 4; i++) async_ld_b128(albase[buf] + i * 16, ap + i * 4);
        const float* wp = wptr + kt * 32;
#pragma unroll
        for (int i = 0; i < 2; i++) async_ld_b128(blbase[buf] + i * 16, wp + i * 4);
    };

    issue(0, 0);
    constexpr int KT = KK / 32;
#pragma unroll 1
    for (int kt = 0; kt < KT; kt++) {
        const int buf = kt & 1;
        wait_async0();          // my tile-kt transfers done
        __syncthreads();        // everyone's transfers done -> LDS tile valid
        if (kt + 1 < KT) issue(kt + 1, buf ^ 1);   // prefetch overlaps compute

        // build fragments (fp32 LDS -> bf16 regs) per ISA 7.12.2 wave32 layouts
        v16bf af[2], bfr[2];
#pragma unroll
        for (int tm = 0; tm < 2; tm++) {
            const float* base = &sA[buf][(wm * 32 + tm * 16 + r) * LDT + kh * 8];
            float4 f0 = *(const float4*)(base);       // K = kh*8 .. +3
            float4 f1 = *(const float4*)(base + 4);   // K = kh*8+4 .. +7
            float4 f2 = *(const float4*)(base + 16);  // K = 16+kh*8 ..
            float4 f3 = *(const float4*)(base + 20);
            union { v16bf v; __bf16 e[16]; } u;
            u.e[0]=(__bf16)f0.x; u.e[1]=(__bf16)f0.y; u.e[2]=(__bf16)f0.z; u.e[3]=(__bf16)f0.w;
            u.e[4]=(__bf16)f1.x; u.e[5]=(__bf16)f1.y; u.e[6]=(__bf16)f1.z; u.e[7]=(__bf16)f1.w;
            u.e[8]=(__bf16)f2.x; u.e[9]=(__bf16)f2.y; u.e[10]=(__bf16)f2.z; u.e[11]=(__bf16)f2.w;
            u.e[12]=(__bf16)f3.x; u.e[13]=(__bf16)f3.y; u.e[14]=(__bf16)f3.z; u.e[15]=(__bf16)f3.w;
            af[tm] = u.v;
        }
#pragma unroll
        for (int tn = 0; tn < 2; tn++) {
            const float* base = &sB[buf][(wn * 32 + tn * 16 + r) * LDT + kh * 8];
            float4 f0 = *(const float4*)(base);       // K = kh*8 .. +3
            float4 f1 = *(const float4*)(base + 4);
            float4 f2 = *(const float4*)(base + 16);
            float4 f3 = *(const float4*)(base + 20);
            union { v16bf v; __bf16 e[16]; } u;
            u.e[0]=(__bf16)f0.x; u.e[1]=(__bf16)f0.y; u.e[2]=(__bf16)f0.z; u.e[3]=(__bf16)f0.w;
            u.e[4]=(__bf16)f1.x; u.e[5]=(__bf16)f1.y; u.e[6]=(__bf16)f1.z; u.e[7]=(__bf16)f1.w;
            u.e[8]=(__bf16)f2.x; u.e[9]=(__bf16)f2.y; u.e[10]=(__bf16)f2.z; u.e[11]=(__bf16)f2.w;
            u.e[12]=(__bf16)f3.x; u.e[13]=(__bf16)f3.y; u.e[14]=(__bf16)f3.z; u.e[15]=(__bf16)f3.w;
            bfr[tn] = u.v;
        }
#pragma unroll
        for (int tm = 0; tm < 2; tm++)
#pragma unroll
            for (int tn = 0; tn < 2; tn++)
                acc[tm][tn] = __builtin_amdgcn_wmma_f32_16x16x32_bf16(
                    false, af[tm], false, bfr[tn], (short)0, acc[tm][tn], false, false);
        __syncthreads();   // done reading buf before it is refilled two steps later
    }

    // epilogue: compile-time NN -> immediate-offset stores
#pragma unroll
    for (int tm = 0; tm < 2; tm++)
#pragma unroll
        for (int tn = 0; tn < 2; tn++) {
            const int row0 = m0 + wm * 32 + tm * 16 + kh * 8;   // rows row0..row0+7 (same batch)
            const int col  = n0 + wn * 32 + tn * 16 + r;
            float* po = Out + (size_t)row0 * NN + col;
            float bv = 0.f, gv = 0.f;
            const float* pr = nullptr;
            if (EPI >= 1) bv = bias[col];
            if (EPI == 2) {
                gv = gate[(size_t)(row0 >> 10) * MODS + col];
                pr = resid + (size_t)row0 * NN + col;
            }
#pragma unroll
            for (int i = 0; i < 8; i++) {
                float v = acc[tm][tn][i];
                if (EPI == 1) {
                    v += bv;
                    float u = 0.7978845608028654f * (v + 0.044715f * v * v * v);
                    v = 0.5f * v * (1.0f + tanhf(u));
                } else if (EPI == 2) {
                    v = pr[i * NN] + gv * (v + bv);
                }
                po[i * NN] = v;
            }
        }
}

// ---- mods = silu(c) @ ada_w.T + ada_b : [8, 6144] ----
__global__ void k_mods(const float* __restrict__ c, const float* __restrict__ ada_w,
                       const float* __restrict__ ada_b, float* __restrict__ mods)
{
    int idx = blockIdx.x * blockDim.x + threadIdx.x;
    if (idx >= BB * MODS) return;
    int bi = idx / MODS, j = idx % MODS;
    const float* cr = c + (size_t)bi * DM;
    const float* wr = ada_w + (size_t)j * DM;
    float s = 0.f;
    for (int k = 0; k < DM; k += 4) {
        float4 cv = *(const float4*)(cr + k);
        float4 wv = *(const float4*)(wr + k);
        s += siluf(cv.x) * wv.x + siluf(cv.y) * wv.y + siluf(cv.z) * wv.z + siluf(cv.w) * wv.w;
    }
    mods[idx] = s + ada_b[j];
}

// ---- layernorm + modulate: one block per row ----
__global__ __launch_bounds__(256) void k_ln_mod(const float* __restrict__ x,
                                                const float* __restrict__ mods,
                                                float* __restrict__ xm,
                                                int sh_off, int sc_off)
{
    __shared__ float red[256];
    const int row = blockIdx.x;
    const int b = row >> 10;
    const float4 v = ((const float4*)(x + (size_t)row * DM))[threadIdx.x];
    red[threadIdx.x] = v.x + v.y + v.z + v.w;
    __syncthreads();
    for (int off = 128; off > 0; off >>= 1) {
        if (threadIdx.x < off) red[threadIdx.x] += red[threadIdx.x + off];
        __syncthreads();
    }
    const float mu = red[0] * (1.0f / DM);
    __syncthreads();
    red[threadIdx.x] = v.x * v.x + v.y * v.y + v.z * v.z + v.w * v.w;
    __syncthreads();
    for (int off = 128; off > 0; off >>= 1) {
        if (threadIdx.x < off) red[threadIdx.x] += red[threadIdx.x + off];
        __syncthreads();
    }
    const float var = red[0] * (1.0f / DM) - mu * mu;
    const float rs = rsqrtf(var + 1e-6f);
    const int c0 = threadIdx.x * 4;
    const float* shp = mods + (size_t)b * MODS + sh_off;
    const float* scp = mods + (size_t)b * MODS + sc_off;
    float4 o;
    o.x = (v.x - mu) * rs * (1.f + scp[c0 + 0]) + shp[c0 + 0];
    o.y = (v.y - mu) * rs * (1.f + scp[c0 + 1]) + shp[c0 + 1];
    o.z = (v.z - mu) * rs * (1.f + scp[c0 + 2]) + shp[c0 + 2];
    o.w = (v.w - mu) * rs * (1.f + scp[c0 + 3]) + shp[c0 + 3];
    ((float4*)(xm + (size_t)row * DM))[threadIdx.x] = o;
}

// ---- depthwise causal conv (DCONV=2) + silu.  xi = xz[:, 0:DI] ----
__global__ void k_conv_silu(const float* __restrict__ xz, const float* __restrict__ conv_w,
                            const float* __restrict__ conv_b, float* __restrict__ x0)
{
    int idx = blockIdx.x * blockDim.x + threadIdx.x;
    if (idx >= MROWS * DI) return;
    int row = idx / DI, d = idx % DI;
    int l = row & 1023;
    float cur  = xz[(size_t)row * (2 * DI) + d];
    float prev = (l == 0) ? 0.f : xz[(size_t)(row - 1) * (2 * DI) + d];
    float v = conv_b[d] + conv_w[d * 2 + 0] * prev + conv_w[d * 2 + 1] * cur;
    x0[idx] = siluf(v);
}

// ---- xdbl = x0 @ xproj_w.T : [8192, 68] ----
__global__ void k_xproj(const float* __restrict__ x0, const float* __restrict__ xproj_w,
                        float* __restrict__ xdbl)
{
    int idx = blockIdx.x * blockDim.x + threadIdx.x;
    if (idx >= MROWS * XPN) return;
    int rrow = idx / XPN, j = idx % XPN;
    const float* a = x0 + (size_t)rrow * DI;
    const float* w = xproj_w + (size_t)j * DI;
    float s = 0.f;
    for (int k = 0; k < DI; k += 4) {
        float4 av = *(const float4*)(a + k);
        float4 wv = *(const float4*)(w + k);
        s += av.x * wv.x + av.y * wv.y + av.z * wv.z + av.w * wv.w;
    }
    xdbl[idx] = s;
}

// ---- dt = softplus(dtr @ dt_w.T + dt_b) : [8192, 2048] ----
__global__ void k_dt(const float* __restrict__ xdbl, const float* __restrict__ dt_w,
                     const float* __restrict__ dt_b, float* __restrict__ dt)
{
    int idx = blockIdx.x * blockDim.x + threadIdx.x;
    if (idx >= MROWS * DI) return;
    int row = idx / DI, d = idx % DI;
    const float* a = xdbl + (size_t)row * XPN;     // cols 0..63 = dtr
    const float* w = dt_w + (size_t)d * DTRN;
    float s = dt_b[d];
    for (int k = 0; k < DTRN; k += 4) {
        float4 av = *(const float4*)(a + k);
        float4 wv = *(const float4*)(w + k);
        s += av.x * wv.x + av.y * wv.y + av.z * wv.z + av.w * wv.w;
    }
    dt[idx] = (s > 20.f) ? s : log1pf(__expf(s));
}

// ---- extract Bm, Cm from xdbl ----
__global__ void k_extract_bc(const float* __restrict__ xdbl, float* __restrict__ Bm,
                             float* __restrict__ Cm)
{
    int idx = blockIdx.x * blockDim.x + threadIdx.x;
    if (idx >= MROWS * DSTATE) return;
    int rr = idx >> 1, n = idx & 1;
    Bm[idx] = xdbl[(size_t)rr * XPN + DTRN + n];
    Cm[idx] = xdbl[(size_t)rr * XPN + DTRN + DSTATE + n];
}

// ---- sequential selective scan + gating: one thread per (b, channel) ----
__global__ __launch_bounds__(256) void k_scan(
    const float* __restrict__ dt, const float* __restrict__ x0,
    const float* __restrict__ Bm, const float* __restrict__ Cm,
    const float* __restrict__ xz,  // z at col offset DI, row stride 2*DI
    const float* __restrict__ A_log, const float* __restrict__ Dv,
    float* __restrict__ y)
{
    int idx = blockIdx.x * blockDim.x + threadIdx.x;
    if (idx >= BB * DI) return;
    int b = idx / DI, d = idx % DI;
    const float A0 = -__expf(A_log[d * 2 + 0]);
    const float A1 = -__expf(A_log[d * 2 + 1]);
    const float Dd = Dv[d];
    float h0 = 0.f, h1 = 0.f;
    for (int l = 0; l < LL; l++) {
        int row = b * LL + l;
        size_t rd = (size_t)row * DI + d;
        float dtv = dt[rd];
        float x0v = x0[rd];
        float du = dtv * x0v;
        float B0 = Bm[row * 2 + 0], B1 = Bm[row * 2 + 1];
        float C0 = Cm[row * 2 + 0], C1 = Cm[row * 2 + 1];
        h0 = __expf(dtv * A0) * h0 + du * B0;
        h1 = __expf(dtv * A1) * h1 + du * B1;
        float yv = h0 * C0 + h1 * C1 + Dd * x0v;
        float zv = xz[(size_t)row * (2 * DI) + DI + d];
        y[rd] = yv * siluf(zv);
    }
}

// ---- x1 = x + g_msa * (out1 + reverse(out2)) ----
__global__ void k_resid1(const float* __restrict__ x, const float* __restrict__ mods,
                         const float* __restrict__ out1, const float* __restrict__ out2,
                         float* __restrict__ x1)
{
    int idx = blockIdx.x * blockDim.x + threadIdx.x;
    if (idx >= MROWS * DM) return;
    int row = idx / DM, col = idx % DM;
    int b = row >> 10, l = row & 1023;
    float g = mods[(size_t)b * MODS + 2 * DM + col];
    float o2 = out2[(size_t)((b << 10) + (1023 - l)) * DM + col];
    x1[idx] = x[idx] + g * (out1[idx] + o2);
}

// ---- aux outputs: A = -exp(A_log1), bias = dt_b1 ----
__global__ void k_aux(const float* __restrict__ A_log1, const float* __restrict__ dt_b1,
                      float* __restrict__ Aout, float* __restrict__ biasout)
{
    int idx = blockIdx.x * blockDim.x + threadIdx.x;
    if (idx < DI * DSTATE) Aout[idx] = -__expf(A_log1[idx]);
    if (idx < DI) biasout[idx] = dt_b1[idx];
}

extern "C" void kernel_launch(void* const* d_in, const int* in_sizes, int n_in,
                              void* d_out, int out_size, void* d_ws, size_t ws_size,
                              hipStream_t stream)
{
    (void)in_sizes; (void)n_in; (void)out_size; (void)ws_size;
    const float* x      = (const float*)d_in[0];
    const float* c      = (const float*)d_in[1];
    const float* mlp_w1 = (const float*)d_in[20];
    const float* mlp_b1 = (const float*)d_in[21];
    const float* mlp_w2 = (const float*)d_in[22];
    const float* mlp_b2 = (const float*)d_in[23];
    const float* ada_w  = (const float*)d_in[24];
    const float* ada_b  = (const float*)d_in[25];

    // output slots: (x, x0, dt, A, Bm, Cm, bias)
    float* out = (float*)d_out;
    float* o_x    = out;
    float* o_x0   = o_x  + (size_t)MROWS * DM;
    float* o_dt   = o_x0 + (size_t)MROWS * DI;
    float* o_A    = o_dt + (size_t)MROWS * DI;
    float* o_Bm   = o_A  + DI * DSTATE;
    float* o_Cm   = o_Bm + MROWS * DSTATE;
    float* o_bias = o_Cm + MROWS * DSTATE;

    // workspace layout (floats)
    float* Wp = (float*)d_ws;
    size_t o = 0;
    float* mods = Wp + o; o += (size_t)BB * MODS;
    float* xm   = Wp + o; o += (size_t)MROWS * DM;      // reused later as xm2
    float* xz   = Wp + o; o += (size_t)MROWS * 2 * DI;  // reused later as h
    float* x0_2 = Wp + o; o += (size_t)MROWS * DI;
    float* dt_2 = Wp + o; o += (size_t)MROWS * DI;
    float* xdbl = Wp + o; o += (size_t)MROWS * XPN;
    float* Bm2  = Wp + o; o += (size_t)MROWS * DSTATE;
    float* Cm2  = Wp + o; o += (size_t)MROWS * DSTATE;
    float* ybuf = Wp + o; o += (size_t)MROWS * DI;
    float* out1 = Wp + o; o += (size_t)MROWS * DM;
    float* out2 = Wp + o; o += (size_t)MROWS * DM;
    float* x1   = Wp + o; o += (size_t)MROWS * DM;
    float* h    = xz;  // reuse

    const dim3 blk(256);

    k_mods<<<dim3((BB * MODS + 255) / 256), blk, 0, stream>>>(c, ada_w, ada_b, mods);
    k_ln_mod<<<dim3(MROWS), blk, 0, stream>>>(x, mods, xm, 0, DM);
    k_aux<<<dim3((DI * DSTATE + 255) / 256), blk, 0, stream>>>(
        (const float*)d_in[8], (const float*)d_in[7], o_A, o_bias);

    // ---- both SSM paths (path 0 = ssm1 forward, path 1 = ssm2 reversed) ----
    for (int p = 0; p < 2; p++) {
        const float* in_w    = (const float*)d_in[2 + 9 * p];
        const float* conv_w  = (const float*)d_in[3 + 9 * p];
        const float* conv_b  = (const float*)d_in[4 + 9 * p];
        const float* xproj_w = (const float*)d_in[5 + 9 * p];
        const float* dt_w    = (const float*)d_in[6 + 9 * p];
        const float* dt_b    = (const float*)d_in[7 + 9 * p];
        const float* A_log   = (const float*)d_in[8 + 9 * p];
        const float* Dv      = (const float*)d_in[9 + 9 * p];
        const float* out_w   = (const float*)d_in[10 + 9 * p];
        float* x0_t = p ? x0_2 : o_x0;
        float* dt_t = p ? dt_2 : o_dt;
        float* Bm_t = p ? Bm2 : o_Bm;
        float* Cm_t = p ? Cm2 : o_Cm;
        float* ot   = p ? out2 : out1;

        dim3 g_xz(2 * DI / 64, MROWS / 128);        // 64 x 64
        if (p == 0)
            gemm_bf16_wmma<0, false, 2 * DI, DM><<<g_xz, blk, 0, stream>>>(
                xm, in_w, nullptr, nullptr, nullptr, xz);
        else
            gemm_bf16_wmma<0, true, 2 * DI, DM><<<g_xz, blk, 0, stream>>>(
                xm, in_w, nullptr, nullptr, nullptr, xz);

        k_conv_silu<<<dim3((MROWS * DI + 255) / 256), blk, 0, stream>>>(xz, conv_w, conv_b, x0_t);
        k_xproj<<<dim3((MROWS * XPN + 255) / 256), blk, 0, stream>>>(x0_t, xproj_w, xdbl);
        k_dt<<<dim3((MROWS * DI + 255) / 256), blk, 0, stream>>>(xdbl, dt_w, dt_b, dt_t);
        k_extract_bc<<<dim3((MROWS * DSTATE + 255) / 256), blk, 0, stream>>>(xdbl, Bm_t, Cm_t);
        k_scan<<<dim3((BB * DI + 255) / 256), blk, 0, stream>>>(
            dt_t, x0_t, Bm_t, Cm_t, xz, A_log, Dv, ybuf);

        dim3 g_out(DM / 64, MROWS / 128);           // 16 x 64
        gemm_bf16_wmma<0, false, DM, DI><<<g_out, blk, 0, stream>>>(
            ybuf, out_w, nullptr, nullptr, nullptr, ot);
    }

    // ---- residual 1 ----
    k_resid1<<<dim3((MROWS * DM + 255) / 256), blk, 0, stream>>>(x, mods, out1, out2, x1);

    // ---- MLP branch ----
    k_ln_mod<<<dim3(MROWS), blk, 0, stream>>>(x1, mods, xm, 3 * DM, 4 * DM);  // xm2
    dim3 g_m1(MLPH / 64, MROWS / 128);
    gemm_bf16_wmma<1, false, MLPH, DM><<<g_m1, blk, 0, stream>>>(
        xm, mlp_w1, mlp_b1, nullptr, nullptr, h);
    dim3 g_m2(DM / 64, MROWS / 128);
    gemm_bf16_wmma<2, false, DM, MLPH><<<g_m2, blk, 0, stream>>>(
        h, mlp_w2, mlp_b2, x1, mods + 5 * DM, o_x);
}
